// MultiViewPrivacyGNN_27212912787885
// MI455X (gfx1250) — matrix operand order repaired and checked by reading, not compiled, not run on hardware
//
#include <hip/hip_runtime.h>
#include <hip/hip_bf16.h>
#include <math.h>

typedef __attribute__((ext_vector_type(16))) __bf16 v16bf;
typedef __attribute__((ext_vector_type(8)))  float  v8f;

__device__ __forceinline__ unsigned short f32_to_bf16(float f) {
    union { float f; unsigned int u; } cv; cv.f = f;
    unsigned int u = cv.u;
    unsigned int r = u + 0x7FFFu + ((u >> 16) & 1u);   // round-to-nearest-even
    return (unsigned short)(r >> 16);
}

// Workgroup barrier that drains ONLY the DS counter (LDS visibility), leaving
// in-flight global loads (tracked by LOADcnt) outstanding across the barrier.
// Same sequence __syncthreads() lowers to, minus the loadcnt drain.
__device__ __forceinline__ void barrier_lds_only() {
    asm volatile("s_wait_dscnt 0x0\n\t"
                 "s_barrier_signal -1\n\t"
                 "s_barrier_wait -1" ::: "memory");
}

// ---------------------------------------------------------------------------
// Kernel 1: per-row L2-normalize (clip 1e-8) + convert to bf16. 1 wave = 1 row.
// ---------------------------------------------------------------------------
__global__ void prep_normalize_bf16(const float* __restrict__ fb,
                                    const float* __restrict__ tw,
                                    unsigned short* __restrict__ fbn,
                                    unsigned short* __restrict__ twn,
                                    int n_fb, int n_tw) {
    int wave = (int)((blockIdx.x * blockDim.x + threadIdx.x) >> 5);
    int lane = threadIdx.x & 31;
    if (wave >= n_fb + n_tw) return;
    const float* src; unsigned short* dst; int row;
    if (wave < n_fb) { src = fb; dst = fbn; row = wave; }
    else             { src = tw; dst = twn; row = wave - n_fb; }
    float x0 = src[row * 64 + lane];
    float x1 = src[row * 64 + 32 + lane];
    float ss = x0 * x0 + x1 * x1;
    #pragma unroll
    for (int m = 16; m >= 1; m >>= 1) ss += __shfl_xor(ss, m, 32);
    float scale = 1.0f / fmaxf(sqrtf(ss), 1e-8f);
    dst[row * 64 + lane]      = f32_to_bf16(x0 * scale);
    dst[row * 64 + 32 + lane] = f32_to_bf16(x1 * scale);
}

// ---------------------------------------------------------------------------
// Kernel 2: fused cosine-sim GEMM (bf16 WMMA, f32 acc) + row argmax.
// Block = 4 waves; each wave owns TWO 16-row fb tiles (block covers 128 rows).
// B tiles (2 KB) are shared through double-buffered LDS. Two-deep software
// pipeline: the global fetch issued at iteration tn is committed to LDS at the
// TOP of iteration tn+1. The inter-iteration barrier waits DScnt only, so the
// staged global load stays in flight across it and L2 latency hides behind a
// full iteration of WMMAs. Argmax is select-form (no EXEC-modifying branches);
// out-of-range columns are clamped duplicates of row n_tw-1 carrying the
// clamped index, so no validity masking is needed and the result is unchanged.
// A/B fragment lane layouts per CDNA5 ISA 7.12.2 (16-bit A 16x32, B 32x16).
// C layout: c[i] at lane l = sims[m][n], m = i + (l>=16)*8, n = l&15.
// ---------------------------------------------------------------------------
__global__ void sim_argmax_wmma(const unsigned short* __restrict__ fbn,
                                const unsigned short* __restrict__ twn,
                                int* __restrict__ bidx,
                                int n_fb, int n_tw) {
    __shared__ alignas(16) unsigned short ldsB[2][16][64];   // double-buffered B tile

    const int lane    = threadIdx.x & 31;
    const int waveBlk = threadIdx.x >> 5;                    // 0..3
    const int tiles_m = (n_fb + 15) >> 4;
    const int tiles_n = (n_tw + 15) >> 4;
    const int tmBase  = (blockIdx.x * 4 + waveBlk) * 2;      // first of 2 M-tiles

    const int halfsel = lane >> 4;                           // 0: lanes 0-15, 1: 16-31
    const int l16     = lane & 15;

    // Cooperative staging map: 128 threads = 16 rows x 8 chunks of 16 bytes.
    const int sRow = threadIdx.x >> 3;                       // 0..15
    const int sChk = (threadIdx.x & 7) * 8;                  // ushort offset, 16B chunks

    union Frag { v16bf v; uint4 q[2]; };

    // --- A fragments (loop-invariant): two 16-row fb tiles, K split 0-31/32-63.
    int rowA0 = tmBase * 16 + l16;       if (rowA0 >= n_fb) rowA0 = n_fb - 1;
    int rowA1 = (tmBase + 1) * 16 + l16; if (rowA1 >= n_fb) rowA1 = n_fb - 1;
    const int abase = halfsel * 8;
    const unsigned short* ap0 = fbn + (size_t)rowA0 * 64 + abase;
    const unsigned short* ap1 = fbn + (size_t)rowA1 * 64 + abase;
    Frag a0A, a1A, a0B, a1B;
    a0A.q[0] = *(const uint4*)(ap0);      a0A.q[1] = *(const uint4*)(ap0 + 16);
    a1A.q[0] = *(const uint4*)(ap0 + 32); a1A.q[1] = *(const uint4*)(ap0 + 48);
    a0B.q[0] = *(const uint4*)(ap1);      a0B.q[1] = *(const uint4*)(ap1 + 16);
    a1B.q[0] = *(const uint4*)(ap1 + 32); a1B.q[1] = *(const uint4*)(ap1 + 48);

    float bv0[8], bv1[8]; int bi0[8], bi1[8];
    #pragma unroll
    for (int i = 0; i < 8; ++i) {
        bv0[i] = -3.4e38f; bv1[i] = -3.4e38f; bi0[i] = 0; bi1[i] = 0;
    }

    // --- Pipeline prologue: stage tile 0 into buffer 0, fetch tile 1 into regs.
    {
        int r = sRow; if (r >= n_tw) r = n_tw - 1;
        *(uint4*)&ldsB[0][sRow][sChk] = *(const uint4*)(twn + (size_t)r * 64 + sChk);
    }
    int rpre = (tiles_n > 1 ? 16 : 0) + sRow; if (rpre >= n_tw) rpre = n_tw - 1;
    uint4 staged = *(const uint4*)(twn + (size_t)rpre * 64 + sChk);
    barrier_lds_only();

    const int bkoff = halfsel * 8;        // B: lanes 0-15 hold K 0-7,16-23; 16-31 hold K 8-15,24-31
    for (int tn = 0; tn < tiles_n; ++tn) {
        const int cur = tn & 1, nxt = cur ^ 1;

        // Commit tile tn+1 (fetched a full iteration ago) to the other buffer.
        *(uint4*)&ldsB[nxt][sRow][sChk] = staged;

        // Issue the fetch of tile tn+2 (clamped) — one iteration to cover latency.
        int tf = tn + 2 < tiles_n ? tn + 2 : tiles_n - 1;
        int rf = tf * 16 + sRow; if (rf >= n_tw) rf = n_tw - 1;
        staged = *(const uint4*)(twn + (size_t)rf * 64 + sChk);

        // Unconditional clamped prefetch four tiles ahead (global_prefetch_b8).
        int tp = tn + 4 < tiles_n ? tn + 4 : tiles_n - 1;
        int rp = tp * 16 + sRow; if (rp >= n_tw) rp = n_tw - 1;
        __builtin_prefetch(twn + (size_t)rp * 64 + sChk, 0, 3);

        // Current B fragments from LDS (shared by both M-tiles).
        Frag b0, b1;
        const unsigned short* lp = &ldsB[cur][l16][bkoff];
        b0.q[0] = *(const uint4*)(lp);
        b0.q[1] = *(const uint4*)(lp + 16);
        b1.q[0] = *(const uint4*)(lp + 32);
        b1.q[1] = *(const uint4*)(lp + 48);

        v8f c0 = {}, c1 = {};
        c0 = __builtin_amdgcn_wmma_f32_16x16x32_bf16(false, a0A.v, false, b0.v,
                                                     (short)0, c0, false, false);
        c0 = __builtin_amdgcn_wmma_f32_16x16x32_bf16(false, a1A.v, false, b1.v,
                                                     (short)0, c0, false, false);
        c1 = __builtin_amdgcn_wmma_f32_16x16x32_bf16(false, a0B.v, false, b0.v,
                                                     (short)0, c1, false, false);
        c1 = __builtin_amdgcn_wmma_f32_16x16x32_bf16(false, a1B.v, false, b1.v,
                                                     (short)0, c1, false, false);

        // Branchless running argmax. Clamped index: phantom tail columns are
        // exact duplicates of row n_tw-1 and carry its index, so no masking.
        int nglob  = tn * 16 + l16;
        int nclamp = nglob < n_tw ? nglob : n_tw - 1;
        #pragma unroll
        for (int i = 0; i < 8; ++i) {
            float s0 = c0[i], s1 = c1[i];
            bool g0 = s0 > bv0[i];
            bool g1 = s1 > bv1[i];
            bv0[i] = g0 ? s0 : bv0[i];
            bi0[i] = g0 ? nclamp : bi0[i];
            bv1[i] = g1 ? s1 : bv1[i];
            bi1[i] = g1 ? nclamp : bi1[i];
        }

        // nxt fully written (DScnt==0) and cur fully consumed; staged global
        // load for tn+2 intentionally remains in flight across the barrier.
        barrier_lds_only();
    }

    // --- Argmax reduction across the 16 column-lanes of each half-wave.
    #pragma unroll
    for (int t = 0; t < 2; ++t) {
        const int tm = tmBase + t;
        #pragma unroll
        for (int i = 0; i < 8; ++i) {
            float v  = t == 0 ? bv0[i] : bv1[i];
            int   bi = t == 0 ? bi0[i] : bi1[i];
            #pragma unroll
            for (int m = 8; m >= 1; m >>= 1) {
                float ov = __shfl_xor(v, m, 32);
                int   oi = __shfl_xor(bi, m, 32);
                if (ov > v || (ov == v && oi < bi)) { v = ov; bi = oi; }
            }
            if (l16 == 0) {
                int row = tm * 16 + halfsel * 8 + i;
                if (row < n_fb) bidx[row] = bi;
            }
        }
    }
}

// ---------------------------------------------------------------------------
// Kernel 3: per-row tail — attention MLP (128->64->2 softmax), weighted sum,
// output MLP (64->32->1 sigmoid). One wave per row, 8 rows per block.
// ---------------------------------------------------------------------------
__global__ void tail_mlp(const float* __restrict__ fb, const float* __restrict__ tw,
                         const int* __restrict__ bidx,
                         const float* __restrict__ Wa1, const float* __restrict__ ba1,
                         const float* __restrict__ Wa2, const float* __restrict__ ba2,
                         const float* __restrict__ Wo1, const float* __restrict__ bo1,
                         const float* __restrict__ Wo2, const float* __restrict__ bo2,
                         float* __restrict__ out, int n_fb) {
    __shared__ float smem[8][128];
    const int lane = threadIdx.x & 31;
    const int w    = threadIdx.x >> 5;
    int row = blockIdx.x * 8 + w;
    int rc  = row < n_fb ? row : n_fb - 1;          // clamp so barriers stay uniform
    int idx = bidx[rc];

    float f0 = fb[(size_t)rc * 64 + lane],  f1 = fb[(size_t)rc * 64 + 32 + lane];
    float t0 = tw[(size_t)idx * 64 + lane], t1 = tw[(size_t)idx * 64 + 32 + lane];
    smem[w][lane]      = f0; smem[w][lane + 32] = f1;
    smem[w][lane + 64] = t0; smem[w][lane + 96] = t1;
    __syncthreads();

    // h = relu(concat @ Wa1 + ba1); lane computes hidden units `lane` and `lane+32`
    float h0 = ba1[lane], h1 = ba1[lane + 32];
    #pragma unroll 4
    for (int p = 0; p < 128; ++p) {
        float cp = smem[w][p];
        h0 = fmaf(cp, Wa1[p * 64 + lane],      h0);
        h1 = fmaf(cp, Wa1[p * 64 + lane + 32], h1);
    }
    h0 = fmaxf(h0, 0.f); h1 = fmaxf(h1, 0.f);

    // 2-way softmax logits via butterfly reduce
    float z0 = h0 * Wa2[2 * lane]     + h1 * Wa2[2 * (lane + 32)];
    float z1 = h0 * Wa2[2 * lane + 1] + h1 * Wa2[2 * (lane + 32) + 1];
    #pragma unroll
    for (int m = 16; m >= 1; m >>= 1) { z0 += __shfl_xor(z0, m, 32); z1 += __shfl_xor(z1, m, 32); }
    z0 += ba2[0]; z1 += ba2[1];
    float mz = fmaxf(z0, z1);
    float e0 = __expf(z0 - mz), e1 = __expf(z1 - mz);
    float a0 = e0 / (e0 + e1), a1 = 1.0f - a0;

    float w0 = a0 * f0 + a1 * t0;
    float w1 = a0 * f1 + a1 * t1;
    __syncthreads();
    smem[w][lane] = w0; smem[w][lane + 32] = w1;
    __syncthreads();

    // h2 = relu(weighted @ Wo1 + bo1); lane computes unit `lane` (32 units)
    float h2 = bo1[lane];
    #pragma unroll 8
    for (int d = 0; d < 64; ++d) h2 = fmaf(smem[w][d], Wo1[d * 32 + lane], h2);
    h2 = fmaxf(h2, 0.f);

    float s = h2 * Wo2[lane];
    #pragma unroll
    for (int m = 16; m >= 1; m >>= 1) s += __shfl_xor(s, m, 32);
    s += bo2[0];
    if (lane == 0 && row < n_fb) out[row] = 1.0f / (1.0f + __expf(-s));
}

// ---------------------------------------------------------------------------
extern "C" void kernel_launch(void* const* d_in, const int* in_sizes, int n_in,
                              void* d_out, int out_size, void* d_ws, size_t ws_size,
                              hipStream_t stream) {
    (void)n_in; (void)out_size; (void)ws_size;
    const float* fb  = (const float*)d_in[0];
    const float* tw  = (const float*)d_in[1];
    const float* Wa1 = (const float*)d_in[2];
    const float* ba1 = (const float*)d_in[3];
    const float* Wa2 = (const float*)d_in[4];
    const float* ba2 = (const float*)d_in[5];
    const float* Wo1 = (const float*)d_in[6];
    const float* bo1 = (const float*)d_in[7];
    const float* Wo2 = (const float*)d_in[8];
    const float* bo2 = (const float*)d_in[9];
    const int n_fb = in_sizes[0] / 64;
    const int n_tw = in_sizes[1] / 64;

    unsigned short* fbn  = (unsigned short*)d_ws;
    unsigned short* twn  = fbn + (size_t)n_fb * 64;
    int*            bidx = (int*)(twn + (size_t)n_tw * 64);
    float*          out  = (float*)d_out;

    const int totalRows = n_fb + n_tw;                       // 1 wave / row, 8 waves / block
    prep_normalize_bf16<<<(totalRows + 7) / 8, 256, 0, stream>>>(fb, tw, fbn, twn, n_fb, n_tw);

    const int tiles_m = (n_fb + 15) / 16;                    // 8 M-tiles per 128-thread block
    sim_argmax_wmma<<<(tiles_m + 7) / 8, 128, 0, stream>>>(fbn, twn, bidx, n_fb, n_tw);

    tail_mlp<<<(n_fb + 7) / 8, 256, 0, stream>>>(fb, tw, bidx, Wa1, ba1, Wa2, ba2,
                                                 Wo1, bo1, Wo2, bo2, out, n_fb);
}
